// GaussianFeatureLeverage_12859132084404
// MI455X (gfx1250) — compile-verified, matrix operand order checked by reading, not run on hardware
//
#include <hip/hip_runtime.h>
#include <hip/hip_bf16.h>
#include <math.h>

typedef _Float16 half_t;
typedef __attribute__((ext_vector_type(16))) _Float16 v16h;
typedef __attribute__((ext_vector_type(8)))  _Float16 v8h;
typedef __attribute__((ext_vector_type(8)))  float    v8f;

static const int kH = 96;
static const int kW = 320;
static const int kN = kH * kW;           // 30720
static const int kCpadFeat = 256;        // 255 feature channels -> 256

__device__ __forceinline__ int imin(int a, int b) { return a < b ? a : b; }
__device__ __forceinline__ int imax(int a, int b) { return a > b ? a : b; }

__device__ __forceinline__ float apply_act(float v, int act) {
  if (act == 1)      v = (v > 0.f) ? v : (expf(v) - 1.f);                    // ELU
  else if (act == 2) v = 0.5f * v * (1.f + erff(v * 0.70710678118654752f));  // exact GELU
  return v;
}

// ---------------------------------------------------------------------------
// Implicit-GEMM 3x3 conv, pad=1 (reflect/edge), stride 1, NHWC f16.
//   src   : f16 [Np][Cpad]    (channel-minor, pad channels zeroed)
//   wpack : f16 [CoutPad][9*Cpad], k = tap*Cpad + c  (zero padded)
//   outH  : f16 [Np][outStride] (pixel-major, pads written as 0)  (optional)
//   outF  : f32 [Cout][Np]      (channel-major, for final outputs) (optional)
// Block 256 threads (8 wave32). Tile 64 Cout x 128 px, K chunk 32 (one tap).
// Double-buffered LDS (1 barrier per chunk). Each wave computes a 2x2 block
// of 16x16 subtiles: 2 A + 2 B operands -> 4 v_wmma_f32_16x16x32_f16.
// ---------------------------------------------------------------------------
__global__ __launch_bounds__(256) void conv3x3_wmma(
    const half_t* __restrict__ src, int Cpad, int Hh, int Ww, int padMode,
    const half_t* __restrict__ wpack, int Cout,
    const float* __restrict__ bias, int act,
    half_t* __restrict__ outH, int outStride,
    float* __restrict__ outF)
{
  __shared__ half_t ldsA[2 * 64 * 32];    // [buf][coutLocal][k]
  __shared__ half_t ldsB[2 * 128 * 32];   // [buf][pixLocal][k]

  const int Np   = Hh * Ww;
  const int Kpad = 9 * Cpad;
  const int tid  = threadIdx.x;
  const int wave = tid >> 5, lane = tid & 31;
  const int M = lane & 15, hb = lane >> 4;

  const int pixTile = blockIdx.x * 128;
  const int coTile  = blockIdx.y * 64;
  const int rp = (wave >> 2) * 2;   // cout subtile row base (0 or 2)
  const int cp = (wave & 3) * 2;    // pixel subtile col base (0,2,4,6)

  // staging: A -> 64 rows x 4 chunks of 8 halves
  const int aRow = tid >> 2, aChunk = tid & 3;
  const half_t* aPtr = wpack + (size_t)(coTile + aRow) * Kpad + aChunk * 8;
  // staging: B -> 128 px x 2 chunks of 16 halves
  const int bPx = tid >> 1, bS = (tid & 1) * 16;
  const int gp = pixTile + bPx;
  const bool pok = gp < Np;
  int gy = 0, gx = 0;
  if (pok) { gy = gp / Ww; gx = gp - gy * Ww; }

  v8f acc[2][2] = {{{}, {}}, {{}, {}}};
  const v8h zero8 = {};
  v8h ra = zero8, rb0 = zero8, rb1 = zero8;

  // ---- prologue: fetch chunk 0 (tap 0 = dy=-1,dx=-1, c0=0), commit to buf 0 ----
  {
    ra = *(const v8h*)(aPtr + 0);
    if (pok) {
      int yy = gy - 1, xx = gx - 1;
      if (padMode == 0) { yy = (yy < 0) ? -yy : yy; xx = (xx < 0) ? -xx : xx; }
      else              { yy = imax(yy, 0);         xx = imax(xx, 0); }
      const half_t* p = src + (size_t)(yy * Ww + xx) * Cpad + bS;
      rb0 = *(const v8h*)p;
      rb1 = *(const v8h*)(p + 8);
    }
  }
  *(v8h*)(&ldsA[aRow * 32 + aChunk * 8]) = ra;
  *(v8h*)(&ldsB[bPx * 32 + bS])     = rb0;
  *(v8h*)(&ldsB[bPx * 32 + bS + 8]) = rb1;
  __syncthreads();

  int tap = 0, cb = 0;   // tap & channel-base of the NEXT chunk
  int cur = 0;
  for (int kb = 0; kb < Kpad; kb += 32) {
    const int kn = kb + 32;
    const bool more = kn < Kpad;

    // ---- issue next chunk's global loads (overlap with WMMA below) ----
    if (more) {
      cb += 32;
      if (cb >= Cpad) { cb = 0; ++tap; }
      ra = *(const v8h*)(aPtr + kn);
      if (kn + 32 < Kpad)
        __builtin_prefetch((const void*)(aPtr + kn + 32), 0, 0);
      if (pok) {
        const int dy = (tap >= 6) ? 1 : ((tap >= 3) ? 0 : -1);
        const int dx = tap - (dy + 1) * 3 - 1;
        int yy = gy + dy, xx = gx + dx;
        if (padMode == 0) {  // reflect
          yy = (yy < 0) ? -yy : ((yy > Hh - 1) ? 2 * (Hh - 1) - yy : yy);
          xx = (xx < 0) ? -xx : ((xx > Ww - 1) ? 2 * (Ww - 1) - xx : xx);
        } else {             // edge
          yy = imin(imax(yy, 0), Hh - 1);
          xx = imin(imax(xx, 0), Ww - 1);
        }
        const half_t* p = src + (size_t)(yy * Ww + xx) * Cpad + cb + bS;
        rb0 = *(const v8h*)p;
        rb1 = *(const v8h*)(p + 8);
      }
    }

    // ---- WMMA from buf[cur]: 2 A operands x 2 B operands -> 4 wmma ----
    // 16-bit operand layout: lane {hb,M}; e<8 -> K=hb*8+e ; e>=8 -> K=16+hb*8+(e-8)
    {
      const half_t* bufA = &ldsA[cur * (64 * 32)];
      const half_t* bufB = &ldsB[cur * (128 * 32)];
      v16h a[2], b[2];
      #pragma unroll
      for (int ri = 0; ri < 2; ++ri) {
        const v8h lo = *(const v8h*)(&bufA[((rp + ri) * 16 + M) * 32 + hb * 8]);
        const v8h hi = *(const v8h*)(&bufA[((rp + ri) * 16 + M) * 32 + 16 + hb * 8]);
        #pragma unroll
        for (int e = 0; e < 8; ++e) { a[ri][e] = lo[e]; a[ri][8 + e] = hi[e]; }
      }
      #pragma unroll
      for (int ci = 0; ci < 2; ++ci) {
        const v8h lo = *(const v8h*)(&bufB[((cp + ci) * 16 + M) * 32 + hb * 8]);
        const v8h hi = *(const v8h*)(&bufB[((cp + ci) * 16 + M) * 32 + 16 + hb * 8]);
        #pragma unroll
        for (int e = 0; e < 8; ++e) { b[ci][e] = lo[e]; b[ci][8 + e] = hi[e]; }
      }
      #pragma unroll
      for (int ri = 0; ri < 2; ++ri)
        #pragma unroll
        for (int ci = 0; ci < 2; ++ci)
          acc[ri][ci] = __builtin_amdgcn_wmma_f32_16x16x32_f16(
              false, a[ri], false, b[ci], (short)0, acc[ri][ci], false, false);
    }

    // ---- commit next chunk to the other buffer; single barrier per chunk ----
    if (more) {
      const int nxt = cur ^ 1;
      *(v8h*)(&ldsA[nxt * (64 * 32) + aRow * 32 + aChunk * 8]) = ra;
      *(v8h*)(&ldsB[nxt * (128 * 32) + bPx * 32 + bS])     = rb0;
      *(v8h*)(&ldsB[nxt * (128 * 32) + bPx * 32 + bS + 8]) = rb1;
      __syncthreads();
      cur = nxt;
    }
  }

  // ---- epilogue: D layout -> lane {hb,M}: row = r + 8*hb, col = M ----
  #pragma unroll
  for (int ri = 0; ri < 2; ++ri) {
    const int coB = (rp + ri) * 16 + 8 * hb;   // local cout base, 8 consecutive rows
    #pragma unroll
    for (int ci = 0; ci < 2; ++ci) {
      const int pn = pixTile + (cp + ci) * 16 + M;
      if (pn >= Np) continue;
      if (outH && (coTile + coB) < outStride) {
        v8h o;
        #pragma unroll
        for (int r = 0; r < 8; ++r) {
          const int co = coTile + coB + r;
          float v = 0.f;
          if (co < Cout) v = apply_act(acc[ri][ci][r] + bias[co], act);
          o[r] = (half_t)v;
        }
        *(v8h*)(&outH[(size_t)pn * outStride + coTile + coB]) = o;
      }
      if (outF) {
        #pragma unroll
        for (int r = 0; r < 8; ++r) {
          const int co = coTile + coB + r;
          if (co < Cout)
            outF[(size_t)co * Np + pn] = apply_act(acc[ri][ci][r] + bias[co], act);
        }
      }
    }
  }
}

// ---------------------------------------------------------------------------
// Weight repack: [Cout][Cin][3][3] f32 -> [CoutPad][9*Cpad] f16, k = tap*Cpad+c
// ---------------------------------------------------------------------------
__global__ void pack_weights_kernel(half_t* __restrict__ dst, const float* __restrict__ w,
                                    int Cout, int Cin, int CoutPad, int Cpad)
{
  const int Kpad = 9 * Cpad;
  int i = blockIdx.x * blockDim.x + threadIdx.x;
  if (i >= CoutPad * Kpad) return;
  int co = i / Kpad, k = i - co * Kpad;
  int tap = k / Cpad, c = k - tap * Cpad;
  float v = (co < Cout && c < Cin) ? w[((size_t)co * Cin + c) * 9 + tap] : 0.f;
  dst[i] = (half_t)v;
}

__global__ void fill_zero_half_kernel(half_t* __restrict__ p, int n)
{
  int i = blockIdx.x * blockDim.x + threadIdx.x;
  if (i < n) p[i] = (half_t)0.f;
}

__global__ void fill_zero_kernel(float* __restrict__ p, int n)
{
  int i = blockIdx.x * blockDim.x + threadIdx.x;
  if (i < n) p[i] = 0.f;
}

// ---------------------------------------------------------------------------
// Bilinear resize (half-pixel centers) f32 [C][hs][ws] -> f16 NHWC [kN][Cpad]
// ---------------------------------------------------------------------------
__global__ void resize_to_f16_kernel(half_t* __restrict__ dst, int Cpad, int chOff,
                                     const float* __restrict__ src,
                                     int C, int hs, int ws)
{
  int i = blockIdx.x * blockDim.x + threadIdx.x;
  if (i >= C * kN) return;
  int c = i / kN, r = i - c * kN;
  int y = r / kW, x = r - y * kW;
  float sy = (y + 0.5f) * hs / (float)kH - 0.5f;
  float sx = (x + 0.5f) * ws / (float)kW - 0.5f;
  sy = fminf(fmaxf(sy, 0.f), (float)(hs - 1));
  sx = fminf(fmaxf(sx, 0.f), (float)(ws - 1));
  int y0 = (int)sy, x0 = (int)sx;
  int y1 = imin(y0 + 1, hs - 1), x1 = imin(x0 + 1, ws - 1);
  float fy = sy - y0, fx = sx - x0;
  const float* s = src + (size_t)c * hs * ws;
  float v = s[y0 * ws + x0] * (1.f - fy) * (1.f - fx) +
            s[y0 * ws + x1] * (1.f - fy) * fx +
            s[y1 * ws + x0] * fy * (1.f - fx) +
            s[y1 * ws + x1] * fy * fx;
  dst[(size_t)r * Cpad + chOff + c] = (half_t)v;
}

// disp = resize(d0), depth = clip(1/(0.01 + 9.99*disp), 0.1, 100)
__global__ void disp_depth_kernel(const float* __restrict__ d0,
                                  float* __restrict__ disp, float* __restrict__ depth)
{
  int i = blockIdx.x * blockDim.x + threadIdx.x;
  if (i >= kN) return;
  const int hs = 24, ws = 80;
  int y = i / kW, x = i - y * kW;
  float sy = (y + 0.5f) * hs / (float)kH - 0.5f;
  float sx = (x + 0.5f) * ws / (float)kW - 0.5f;
  sy = fminf(fmaxf(sy, 0.f), (float)(hs - 1));
  sx = fminf(fmaxf(sx, 0.f), (float)(ws - 1));
  int y0 = (int)sy, x0 = (int)sx;
  int y1 = imin(y0 + 1, hs - 1), x1 = imin(x0 + 1, ws - 1);
  float fy = sy - y0, fx = sx - x0;
  float v = d0[y0 * ws + x0] * (1.f - fy) * (1.f - fx) +
            d0[y0 * ws + x1] * (1.f - fy) * fx +
            d0[y1 * ws + x0] * fy * (1.f - fx) +
            d0[y1 * ws + x1] * fy * fx;
  disp[i] = v;
  float dep = 1.f / (0.01f + 9.99f * v);
  depth[i] = fminf(fmaxf(dep, 0.1f), 100.f);
}

// ---------------------------------------------------------------------------
// Gaussian generation; gp is f16 pixel-major [kN][16]:
// ch [0..2]=off, [3]=opa, [4..6]=scl ([7..10]=rot unused by rasterize -> dead)
// ---------------------------------------------------------------------------
__global__ void gs_generator_kernel(const half_t* __restrict__ gp16,
                                    const float* __restrict__ depth,
                                    const float* __restrict__ disp, int addDisp,
                                    const float* __restrict__ iK,
                                    float* __restrict__ pos, float* __restrict__ scl,
                                    float* __restrict__ opa)
{
  int n = blockIdx.x * blockDim.x + threadIdx.x;
  if (n >= kN) return;
  const half_t* g = gp16 + (size_t)n * 16;
  float x = (float)(n % kW), y = (float)(n / kW);
  float dpt = depth[n] + (addDisp ? disp[n] : 0.f);
  float camx = iK[0] * x + iK[1] * y + iK[2];
  float camy = iK[4] * x + iK[5] * y + iK[6];
  float camz = iK[8] * x + iK[9] * y + iK[10];
  pos[n * 3 + 0] = camx * dpt + (float)g[0];
  pos[n * 3 + 1] = camy * dpt + (float)g[1];
  pos[n * 3 + 2] = camz * dpt + (float)g[2];
  opa[n] = 1.f / (1.f + expf(-(float)g[3]));
  scl[n * 3 + 0] = expf((float)g[4]) * 0.01f;
  scl[n * 3 + 1] = expf((float)g[5]) * 0.01f;
  scl[n * 3 + 2] = expf((float)g[6]) * 0.01f;
}

// ---------------------------------------------------------------------------
// Scatter splat; gfe is f16 pixel-major [kN][64] (contiguous per gaussian)
// num: [P][64] f32, den: [P] f32
// ---------------------------------------------------------------------------
__global__ void rasterize_kernel(const float* __restrict__ pos, const float* __restrict__ scl,
                                 const float* __restrict__ opa, const half_t* __restrict__ gfe,
                                 const float* __restrict__ Km, int hs, int ws,
                                 float* __restrict__ num, float* __restrict__ den)
{
  int n = blockIdx.x * blockDim.x + threadIdx.x;
  if (n >= kN) return;
  float x = pos[n * 3 + 0], y = pos[n * 3 + 1], z = pos[n * 3 + 2];
  float fx = Km[0], cx = Km[2], fy = Km[5], cy = Km[6];
  float zc = fmaxf(z, 1e-3f);
  float u = fx * x / zc + cx;
  float v = fy * y / zc + cy;
  bool valid = (z > 0.1f) && (z < 100.f) &&
               (u >= 0.f) && (u <= (float)(ws - 1)) &&
               (v >= 0.f) && (v <= (float)(hs - 1));
  if (!valid) return;
  float sm = (scl[n * 3] + scl[n * 3 + 1] + scl[n * 3 + 2]) * (1.f / 3.f);
  float sigma = fmaxf(fx * sm / zc, 0.3f);
  float u0 = floorf(u), v0 = floorf(v);
  float op = opa[n];
  const half_t* fsrc = gfe + (size_t)n * 64;
  #pragma unroll
  for (int d = 0; d < 4; ++d) {
    float uc = u0 + (float)(d & 1);
    float vc = v0 + (float)(d >> 1);
    if (uc > (float)(ws - 1) || vc > (float)(hs - 1)) continue;
    float du = u - uc, dv = v - vc;
    float g = expf(-0.5f * (du * du + dv * dv) / (sigma * sigma));
    float w = op * g;
    int ix = imin(imax((int)uc, 0), ws - 1);
    int iy = imin(imax((int)vc, 0), hs - 1);
    int idx = iy * ws + ix;
    atomicAdd(&den[idx], w);
    float* nrow = num + (size_t)idx * 64;
    for (int c = 0; c < 64; ++c)
      atomicAdd(&nrow[c], w * (float)fsrc[c]);
  }
}

// fused[p*CfPad + chBase + c] = num[p*64+c] / (den[p]+1e-8)
__global__ void norm_to_fused_kernel(half_t* __restrict__ fused, int CfPad,
                                     const float* __restrict__ num,
                                     const float* __restrict__ den, int P, int chBase)
{
  int i = blockIdx.x * blockDim.x + threadIdx.x;
  if (i >= 64 * P) return;
  int p = i >> 6, c = i & 63;
  fused[(size_t)p * CfPad + chBase + c] =
      (half_t)(num[(size_t)p * 64 + c] / (den[p] + 1e-8f));
}

// channel-major f32 [C][P] -> fused NHWC f16 [P][CfPad] at channel offset
__global__ void cast_cm_to_fused_kernel(half_t* __restrict__ fused, int CfPad, int chOff,
                                        const float* __restrict__ src, int C, int P)
{
  int i = blockIdx.x * blockDim.x + threadIdx.x;
  if (i >= C * P) return;
  int c = i / P, p = i - c * P;
  fused[(size_t)p * CfPad + chOff + c] = (half_t)src[i];
}

// ---------------------------------------------------------------------------
// Host launcher
// ---------------------------------------------------------------------------
static inline int round_up(int v, int m) { return ((v + m - 1) / m) * m; }

extern "C" void kernel_launch(void* const* d_in, const int* in_sizes, int n_in,
                              void* d_out, int out_size, void* d_ws, size_t ws_size,
                              hipStream_t stream)
{
  (void)in_sizes; (void)n_in; (void)out_size; (void)ws_size;

  // ---- input map (setup_inputs dict order; params pytree key-sorted) ----
  const float* f[4]  = {(const float*)d_in[0], (const float*)d_in[3],
                        (const float*)d_in[6], (const float*)d_in[9]};
  const float* dd[4] = {(const float*)d_in[1], (const float*)d_in[4],
                        (const float*)d_in[7], (const float*)d_in[10]};
  const float* cc[4] = {(const float*)d_in[2], (const float*)d_in[5],
                        (const float*)d_in[8], (const float*)d_in[11]};  // c2..c5
  const float* c0   = (const float*)d_in[12];
  const float* invK = (const float*)d_in[13];
  const float* Km[4] = {(const float*)d_in[14], (const float*)d_in[15],
                        (const float*)d_in[16], (const float*)d_in[17]};
  const float* hb1[2] = {(const float*)d_in[18], (const float*)d_in[22]};
  const float* hb2[2] = {(const float*)d_in[19], (const float*)d_in[23]};
  const float* hw1[2] = {(const float*)d_in[20], (const float*)d_in[24]};
  const float* hw2[2] = {(const float*)d_in[21], (const float*)d_in[25]};
  const float* lb1[2] = {(const float*)d_in[26], (const float*)d_in[30]};
  const float* lb2[2] = {(const float*)d_in[27], (const float*)d_in[31]};
  const float* lw1[2] = {(const float*)d_in[28], (const float*)d_in[32]};
  const float* lw2[2] = {(const float*)d_in[29], (const float*)d_in[33]};
  const float* rb[4] = {(const float*)d_in[34], (const float*)d_in[36],
                        (const float*)d_in[38], (const float*)d_in[40]};
  const float* rw[4] = {(const float*)d_in[35], (const float*)d_in[37],
                        (const float*)d_in[39], (const float*)d_in[41]};

  const int chIn[4]  = {24, 32, 64, 128};
  const int chOut[4] = {18, 36, 72, 144};
  const int hsL[4] = {24, 12, 6, 3};
  const int wsL[4] = {80, 40, 20, 10};

  // ---- workspace carve-out ----
  char* wsp = (char*)d_ws;
  auto alloc = [&](size_t bytes) -> void* {
    void* p = (void*)wsp;
    wsp += (bytes + 255) & ~(size_t)255;
    return p;
  };
  half_t* featH  = (half_t*)alloc((size_t)kN * kCpadFeat * 2);
  float*  dispB  = (float*) alloc((size_t)kN * 4);
  float*  depthB = (float*) alloc((size_t)kN * 4);

  half_t* pHW1[2]; half_t* pHW2[2]; half_t* pLW1[2]; half_t* pLW2[2];
  for (int i = 0; i < 2; ++i) {
    pHW1[i] = (half_t*)alloc((size_t)64  * 9 * kCpadFeat * 2);
    pHW2[i] = (half_t*)alloc((size_t)64  * 9 * 32 * 2);
    pLW1[i] = (half_t*)alloc((size_t)256 * 9 * kCpadFeat * 2);
    pLW2[i] = (half_t*)alloc((size_t)64  * 9 * kCpadFeat * 2);
  }
  int Cf[4], CfPad[4], CoPadRes[4];
  half_t* pRW[4];
  for (int i = 0; i < 4; ++i) {
    Cf[i]    = 128 + chIn[i] + 4;
    CfPad[i] = round_up(Cf[i], 32);
    CoPadRes[i] = round_up(chOut[i], 64);
    pRW[i] = (half_t*)alloc((size_t)CoPadRes[i] * 9 * CfPad[i] * 2);
  }
  half_t* h1buf  = (half_t*)alloc((size_t)kN * 32 * 2);   // [kN][32], Cout=22
  half_t* gpbuf  = (half_t*)alloc((size_t)kN * 16 * 2);   // [kN][16], Cout=11
  half_t* tmp255 = (half_t*)alloc((size_t)kN * kCpadFeat * 2);
  half_t* gfeH[2]; float* posB[2]; float* sclB[2]; float* opaB[2];
  for (int i = 0; i < 2; ++i) {
    gfeH[i] = (half_t*)alloc((size_t)kN * 64 * 2);        // [kN][64]
    posB[i] = (float*)alloc((size_t)3 * kN * 4);
    sclB[i] = (float*)alloc((size_t)3 * kN * 4);
    opaB[i] = (float*)alloc((size_t)kN * 4);
  }
  const int Pmax = 24 * 80;
  float* numB[2]; float* denB[2];
  for (int j = 0; j < 2; ++j) {
    numB[j] = (float*)alloc((size_t)Pmax * 64 * 4);
    denB[j] = (float*)alloc((size_t)Pmax * 4);
  }
  half_t* fused = (half_t*)alloc((size_t)Pmax * 288 * 2);

  float* outBase = (float*)d_out;
  const int outOff[4] = {0, 18 * 1920, 18 * 1920 + 36 * 480,
                         18 * 1920 + 36 * 480 + 72 * 120};

  auto gsz = [](int n) { return (n + 255) / 256; };
  auto launch_pack = [&](half_t* dst, const float* w, int Cout, int Cin, int Cpad) {
    int CoutPad = round_up(Cout, 64);
    int total = CoutPad * 9 * Cpad;
    pack_weights_kernel<<<gsz(total), 256, 0, stream>>>(dst, w, Cout, Cin, CoutPad, Cpad);
  };
  auto launch_conv = [&](const half_t* src, int Cpad, int Hh, int Ww, int pad,
                         const half_t* wp, int Cout, const float* bias, int act,
                         half_t* oH, int oStride, float* oF) {
    int Np = Hh * Ww;
    dim3 grid((Np + 127) / 128, (Cout + 63) / 64);
    conv3x3_wmma<<<grid, 256, 0, stream>>>(src, Cpad, Hh, Ww, pad, wp, Cout,
                                           bias, act, oH, oStride, oF);
  };

  // ---- weight repacks ----
  for (int i = 0; i < 2; ++i) {
    launch_pack(pHW1[i], hw1[i], 22, 255, kCpadFeat);
    launch_pack(pHW2[i], hw2[i], 11, 22, 32);
    launch_pack(pLW1[i], lw1[i], 255, 255, kCpadFeat);
    launch_pack(pLW2[i], lw2[i], 64, 255, kCpadFeat);
  }
  for (int i = 0; i < 4; ++i) launch_pack(pRW[i], rw[i], chOut[i], Cf[i], CfPad[i]);

  // ---- build concatenated feature NHWC f16 (zero pad channel), disp/depth ----
  fill_zero_half_kernel<<<gsz(kN * kCpadFeat), 256, 0, stream>>>(featH, kN * kCpadFeat);
  int chOff = 0;
  for (int i = 0; i < 4; ++i) {
    resize_to_f16_kernel<<<gsz(chIn[i] * kN), 256, 0, stream>>>(
        featH, kCpadFeat, chOff, f[i], chIn[i], hsL[i], wsL[i]);
    chOff += chIn[i];
  }
  for (int i = 0; i < 4; ++i)
    resize_to_f16_kernel<<<gsz(kN), 256, 0, stream>>>(
        featH, kCpadFeat, 248 + i, dd[i], 1, hsL[i], wsL[i]);
  resize_to_f16_kernel<<<gsz(3 * kN), 256, 0, stream>>>(
      featH, kCpadFeat, 252, c0, 3, kH, kW);  // identity resize
  disp_depth_kernel<<<gsz(kN), 256, 0, stream>>>(dd[0], dispB, depthB);

  // ---- per GS branch: head convs, gaussian gen, lev convs ----
  for (int i = 0; i < 2; ++i) {
    launch_conv(featH, kCpadFeat, kH, kW, /*edge*/1, pHW1[i], 22, hb1[i], /*GELU*/2,
                h1buf, 32, nullptr);
    launch_conv(h1buf, 32, kH, kW, /*edge*/1, pHW2[i], 11, hb2[i], /*none*/0,
                gpbuf, 16, nullptr);
    gs_generator_kernel<<<gsz(kN), 256, 0, stream>>>(gpbuf, depthB, dispB, i,
                                                     invK, posB[i], sclB[i], opaB[i]);
    launch_conv(featH, kCpadFeat, kH, kW, /*reflect*/0, pLW1[i], 255, lb1[i], /*ELU*/1,
                tmp255, kCpadFeat, nullptr);
    launch_conv(tmp255, kCpadFeat, kH, kW, /*reflect*/0, pLW2[i], 64, lb2[i], /*ELU*/1,
                gfeH[i], 64, nullptr);
  }

  // ---- per level: rasterize both branches, fuse, final conv ----
  for (int i = 0; i < 4; ++i) {
    const int hs = hsL[i], ws = wsL[i], P = hs * ws;
    fill_zero_half_kernel<<<gsz(P * CfPad[i]), 256, 0, stream>>>(fused, P * CfPad[i]);
    for (int j = 0; j < 2; ++j) {
      fill_zero_kernel<<<gsz(P * 64), 256, 0, stream>>>(numB[j], P * 64);
      fill_zero_kernel<<<gsz(P), 256, 0, stream>>>(denB[j], P);
    }
    for (int j = 0; j < 2; ++j)
      rasterize_kernel<<<gsz(kN), 256, 0, stream>>>(posB[j], sclB[j], opaB[j],
                                                    gfeH[j], Km[i], hs, ws,
                                                    numB[j], denB[j]);
    for (int j = 0; j < 2; ++j)
      norm_to_fused_kernel<<<gsz(64 * P), 256, 0, stream>>>(fused, CfPad[i],
                                                            numB[j], denB[j], P, 64 * j);
    cast_cm_to_fused_kernel<<<gsz(chIn[i] * P), 256, 0, stream>>>(
        fused, CfPad[i], 128, f[i], chIn[i], P);
    cast_cm_to_fused_kernel<<<gsz(3 * P), 256, 0, stream>>>(
        fused, CfPad[i], 128 + chIn[i], cc[i], 3, P);
    cast_cm_to_fused_kernel<<<gsz(P), 256, 0, stream>>>(
        fused, CfPad[i], 128 + chIn[i] + 3, dd[i], 1, P);
    launch_conv(fused, CfPad[i], hs, ws, /*reflect*/0, pRW[i], chOut[i], rb[i], /*ELU*/1,
                nullptr, 0, outBase + outOff[i]);
  }
}
